// Model_28252294873752
// MI455X (gfx1250) — matrix operand order repaired and checked by reading, not compiled, not run on hardware
//
#include <hip/hip_runtime.h>
#include <hip/hip_bf16.h>

typedef __attribute__((ext_vector_type(16))) __bf16 v16bf;
typedef __attribute__((ext_vector_type(8)))  float  v8f;

#define N_NODES 50000

__device__ __forceinline__ float bf_lo(unsigned u) { return __uint_as_float(u << 16); }
__device__ __forceinline__ float bf_hi(unsigned u) { return __uint_as_float(u & 0xffff0000u); }

// ---------------------------------------------------------------------------
// Degree / normalization
// ---------------------------------------------------------------------------
__global__ void deg_init(float* __restrict__ deg, int n) {
    int i = blockIdx.x * blockDim.x + threadIdx.x;
    if (i < n) deg[i] = 1.0f;  // self-loop contribution
}

__global__ void deg_count(const int* __restrict__ dst, float* __restrict__ deg, int e) {
    int i = blockIdx.x * blockDim.x + threadIdx.x;
    if (i < e) atomicAdd(&deg[dst[i]], 1.0f);
}

__global__ void deg_rsqrt(float* __restrict__ deg, int n) {
    int i = blockIdx.x * blockDim.x + threadIdx.x;
    if (i < n) deg[i] = rsqrtf(deg[i]);  // deg >= 1 always (self-loop)
}

// ---------------------------------------------------------------------------
// GEMM: Out[N,F] (bf16) = A[N,K] (f32 -> bf16) @ W[K,F] (f32 -> bf16),
// f32 accumulate in registers, bf16 store (halves downstream gather traffic).
// One wave per 16x16 output tile; v_wmma_f32_16x16x32_bf16 over K in steps of 32.
// VGPR layouts per CDNA5 ISA 7.12.2 (wave32).
// ---------------------------------------------------------------------------
template <int K, int F>
__global__ void gemm_bf16_wmma(const float* __restrict__ A,
                               const float* __restrict__ W,
                               __bf16* __restrict__ Out,
                               int nRowTiles) {
    const int colTiles = F / 16;
    int wave = blockIdx.x * (blockDim.x >> 5) + (threadIdx.x >> 5);
    int lane = threadIdx.x & 31;
    int rowTile = wave / colTiles;       // wave-uniform
    int colTile = wave % colTiles;       // wave-uniform
    if (rowTile >= nRowTiles) return;    // wave-uniform branch: EXEC stays all-1s

    int m   = lane & 15;                 // A-row / B-col index within tile
    int hlf = lane >> 4;                 // lane-half selector

    v8f acc = {};
#pragma unroll
    for (int kb = 0; kb < K; kb += 32) {
        // A: 16x32 bf16. lanes 0-15: a[0..7]=K(kb+0..7),   a[8..15]=K(kb+16..23)
        //                lanes16-31: a[0..7]=K(kb+8..15),  a[8..15]=K(kb+24..31)
        v16bf a;
        const float* ap = A + (size_t)(rowTile * 16 + m) * K + kb;
#pragma unroll
        for (int j = 0; j < 8; ++j) {
            a[j]     = (__bf16)ap[hlf * 8 + j];
            a[8 + j] = (__bf16)ap[16 + hlf * 8 + j];
        }
        // B: 32x16 bf16. b[0..7]=K(kb+hlf*8+0..7), b[8..15]=K(kb+16+hlf*8+0..7), col = m
        v16bf b;
        const float* wp = W + (size_t)(kb + hlf * 8) * F + colTile * 16 + m;
#pragma unroll
        for (int j = 0; j < 16; ++j) {
            b[j] = (__bf16)wp[(size_t)(j < 8 ? j : j + 8) * F];
        }
        acc = __builtin_amdgcn_wmma_f32_16x16x32_bf16(
            /*neg_a=*/false, a, /*neg_b=*/false, b,
            /*c_mod=*/(short)0, acc, /*reuse_a=*/false, /*reuse_b=*/false);
    }

    // C/D: acc[v] = D[row = hlf*8 + v, col = m] within the tile; store bf16
    __bf16* op = Out + (size_t)(rowTile * 16 + hlf * 8) * F + colTile * 16 + m;
#pragma unroll
    for (int v = 0; v < 8; ++v) op[(size_t)v * F] = (__bf16)acc[v];
}

// ---------------------------------------------------------------------------
// Aggregation init: out (f32) = dis[i]^2 * h[i][c]  (self-loop term; full
// overwrite each call -> graph-replay safe)
// ---------------------------------------------------------------------------
template <int LOGF>
__global__ void self_agg_init(const __bf16* __restrict__ h,
                              const float* __restrict__ dis,
                              float* __restrict__ out, int n) {
    int i = blockIdx.x * blockDim.x + threadIdx.x;
    if (i >= n) return;
    int node = i >> LOGF;
    float d = dis[node];
    out[i] = d * d * (float)h[i];
}

// One wave per edge; each lane owns F/32 contiguous channels.
// h is bf16 -> 16B (F=256) or 8B (F=128) per lane, expanded to f32 via shifts;
// accumulation stays in f32 global atomics for segment-sum precision.
template <int F, int CPL>  // CPL = F/32 channels per lane
__global__ void edge_scatter(const __bf16* __restrict__ h,
                             const int* __restrict__ src,
                             const int* __restrict__ dst,
                             const float* __restrict__ dis,
                             float* __restrict__ out, int nEdges) {
    int t = blockIdx.x * blockDim.x + threadIdx.x;
    int e = t >> 5;
    int lane = t & 31;
    if (e >= nEdges) return;
    int s = src[e];
    int d = dst[e];
    float norm = dis[s] * dis[d];
    const unsigned* hp =
        (const unsigned*)(h + (size_t)s * F) + lane * (CPL / 2);
    float* op = out + (size_t)d * F + lane * CPL;
#pragma unroll
    for (int j = 0; j < CPL / 2; ++j) {
        unsigned u = hp[j];
        atomicAdd(op + 2 * j + 0, bf_lo(u) * norm);
        atomicAdd(op + 2 * j + 1, bf_hi(u) * norm);
    }
}

template <int LOGF>
__global__ void bias_relu(float* __restrict__ a, const float* __restrict__ bias, int n) {
    int i = blockIdx.x * blockDim.x + threadIdx.x;
    if (i >= n) return;
    float v = a[i] + bias[i & ((1 << LOGF) - 1)];
    a[i] = v > 0.0f ? v : 0.0f;
}

// ---------------------------------------------------------------------------
extern "C" void kernel_launch(void* const* d_in, const int* in_sizes, int n_in,
                              void* d_out, int out_size, void* d_ws, size_t ws_size,
                              hipStream_t stream) {
    const float* x    = (const float*)d_in[0];
    const int*   edge = (const int*)d_in[1];   // [2, E] flattened
    // d_in[2] = batch (unused)
    const float* W1 = (const float*)d_in[3];
    const float* b1 = (const float*)d_in[4];
    const float* W2 = (const float*)d_in[5];
    const float* b2 = (const float*)d_in[6];

    const int N = N_NODES;
    const int E = in_sizes[1] / 2;             // 800000
    const int* src = edge;
    const int* dst = edge + E;

    float*  ws   = (float*)d_ws;
    float*  dis  = ws;                         // [N] f32
    __bf16* hbuf = (__bf16*)(ws + N);          // [N*256] bf16 (h1, later h2)
    float*  aggr = ws + N + (size_t)N * 128;   // [N*256] f32  (agg1 -> a1)

    float* out = (float*)d_out;                // [N*128] f32

    const int TPB = 256;

    // --- normalization ---
    deg_init<<<(N + TPB - 1) / TPB, TPB, 0, stream>>>(dis, N);
    deg_count<<<(E + TPB - 1) / TPB, TPB, 0, stream>>>(dst, dis, E);
    deg_rsqrt<<<(N + TPB - 1) / TPB, TPB, 0, stream>>>(dis, N);

    // --- layer 1: h1 = x @ W1 (bf16) ; agg (f32) ; relu(+b1) ---
    {
        const int rowTiles = N / 16;           // 3125 exactly
        const int waves = rowTiles * (256 / 16);
        const int blocks = (waves + 7) / 8;    // 8 waves / 256-thread block
        gemm_bf16_wmma<128, 256><<<blocks, TPB, 0, stream>>>(x, W1, hbuf, rowTiles);

        int n1 = N * 256;
        self_agg_init<8><<<(n1 + TPB - 1) / TPB, TPB, 0, stream>>>(hbuf, dis, aggr, n1);
        long long th = (long long)E * 32;
        edge_scatter<256, 8><<<(int)((th + TPB - 1) / TPB), TPB, 0, stream>>>(
            hbuf, src, dst, dis, aggr, E);
        bias_relu<8><<<(n1 + TPB - 1) / TPB, TPB, 0, stream>>>(aggr, b1, n1);
    }

    // --- layer 2: h2 = a1 @ W2 (bf16) ; agg into d_out ; relu(+b2) ---
    {
        const int rowTiles = N / 16;
        const int waves = rowTiles * (128 / 16);
        const int blocks = (waves + 7) / 8;
        gemm_bf16_wmma<256, 128><<<blocks, TPB, 0, stream>>>(aggr, W2, hbuf, rowTiles);

        int n2 = N * 128;
        self_agg_init<7><<<(n2 + TPB - 1) / TPB, TPB, 0, stream>>>(hbuf, dis, out, n2);
        long long th = (long long)E * 32;
        edge_scatter<128, 4><<<(int)((th + TPB - 1) / TPB), TPB, 0, stream>>>(
            hbuf, src, dst, dis, out, E);
        bias_relu<7><<<(n2 + TPB - 1) / TPB, TPB, 0, stream>>>(out, b2, n2);
    }
}
